// get_model_20160576487516
// MI455X (gfx1250) — matrix operand order repaired and checked by reading, not compiled, not run on hardware
//
#include <hip/hip_runtime.h>

typedef __attribute__((ext_vector_type(16))) _Float16 v16h;
typedef __attribute__((ext_vector_type(8)))  float    v8f;

#define BATCH 8
#define NPTS  4096
#define DE    128
#define SPTS  1024
#define KNN   32
#define FLATF 4096   // 32*128
#define EPS_BN 1e-5f

// ---------------- WMMA fragment helpers (this code's contiguous k order, shared by A and B) ----------------
// A 16xK (16-bit): lane L holds row M=L&15; K = k0 + ((L>=16)?16:0) + [0,16)
__device__ __forceinline__ v16h frag_ld_a(const _Float16* base, int stride, int row0, int k0) {
  int lane = threadIdx.x & 31;
  const _Float16* p = base + (size_t)(row0 + (lane & 15)) * stride + k0 + ((lane >> 4) << 4);
  v16h f;
#pragma unroll
  for (int i = 0; i < 16; ++i) f[i] = p[i];
  return f;
}

// B 32x16 (16-bit): lane L holds column N=L&15, K = k0 + ((L>=16)?16:0) + [0,16).
// Memory holds B^T (N-major rows, e.g. weight (out,in)) -> row reads.
__device__ __forceinline__ v16h frag_ld_bt(const _Float16* base, int stride, int col0, int k0) {
  int lane = threadIdx.x & 31;
  const _Float16* p = base + (size_t)(col0 + (lane & 15)) * stride + k0 + ((lane >> 4) << 4);
  v16h f;
#pragma unroll
  for (int i = 0; i < 16; ++i) f[i] = p[i];
  return f;
}

// Memory holds B directly (K-major rows, K x N) -> column reads.
__device__ __forceinline__ v16h frag_ld_b(const _Float16* base, int stride, int col0, int k0) {
  int lane = threadIdx.x & 31;
  int n = col0 + (lane & 15);
  int kb = k0 + ((lane >> 4) << 4);
  v16h f;
#pragma unroll
  for (int i = 0; i < 16; ++i) f[i] = base[(size_t)(kb + i) * stride + n];
  return f;
}

__device__ __forceinline__ v8f wmma32(v16h a, v16h b, v8f c) {
  return __builtin_amdgcn_wmma_f32_16x16x32_f16(false, a, false, b, (short)0, c, false, false);
}

// ---------------- CDNA5 async global->LDS copy (ASYNCcnt path) ----------------
__device__ __forceinline__ void async_b128(_Float16* lds_dst, const _Float16* gsrc) {
  unsigned lds = (unsigned)(uintptr_t)lds_dst;                 // low 32 bits = LDS address
  unsigned long long ga = (unsigned long long)(uintptr_t)gsrc; // 64-bit global address
  asm volatile("global_load_async_to_lds_b128 %0, %1, off" :: "v"(lds), "v"(ga) : "memory");
}
__device__ __forceinline__ void wait_async0() {
  asm volatile("s_wait_asynccnt 0x0" ::: "memory");
}

// ---------------- small convert ----------------
__global__ __launch_bounds__(256) void cvt_f16_kernel(const float* __restrict__ src,
                                                      _Float16* __restrict__ dst, int n) {
  int i = blockIdx.x * 256 + threadIdx.x;
  if (i < n) dst[i] = (_Float16)src[i];
}

// ---------------- emb MLP: (B*N,6) -> (B*N,128) ----------------
__global__ __launch_bounds__(256) void emb_kernel(
    const float* __restrict__ xin,
    const float* __restrict__ w1, const float* __restrict__ b1,
    const float* __restrict__ w2, const float* __restrict__ b2,
    const float* __restrict__ g1, const float* __restrict__ bb1,
    const float* __restrict__ m1, const float* __restrict__ vv1,
    const float* __restrict__ g2, const float* __restrict__ bb2,
    const float* __restrict__ m2, const float* __restrict__ vv2,
    float* __restrict__ points)
{
  int idx = blockIdx.x * 256 + threadIdx.x;
  if (idx >= BATCH * NPTS) return;
  float xi[6];
#pragma unroll
  for (int j = 0; j < 6; ++j) xi[j] = xin[(size_t)idx * 6 + j];
  float h[64];
  for (int c = 0; c < 64; ++c) {
    float a = b1[c];
#pragma unroll
    for (int j = 0; j < 6; ++j) a += w1[c * 6 + j] * xi[j];
    a = (a - m1[c]) * rsqrtf(vv1[c] + EPS_BN) * g1[c] + bb1[c];
    h[c] = fmaxf(a, 0.f);
  }
  float* o = points + (size_t)idx * DE;
  for (int c = 0; c < DE; ++c) {
    float a = b2[c];
    for (int j = 0; j < 64; ++j) a += w2[c * 64 + j] * h[j];
    a = (a - m2[c]) * rsqrtf(vv2[c] + EPS_BN) * g2[c] + bb2[c];
    o[c] = fmaxf(a, 0.f);
  }
}

// ---------------- farthest point sampling: 1 block per batch ----------------
__global__ __launch_bounds__(1024) void fps_kernel(
    const float* __restrict__ xin, int* __restrict__ fps_idx, float* __restrict__ center)
{
  int b = blockIdx.x, tid = threadIdx.x;
  __shared__ float dist[NPTS];
  __shared__ float rv[1024];
  __shared__ int   ri[1024];
  __shared__ float cc[3];
  for (int i = tid; i < NPTS; i += 1024) dist[i] = 1e10f;
  int far = 0;
  __syncthreads();
  for (int it = 0; it < SPTS; ++it) {
    if (tid == 0) {
      fps_idx[b * SPTS + it] = far;
      const float* p = xin + ((size_t)b * NPTS + far) * 6;
      cc[0] = p[0]; cc[1] = p[1]; cc[2] = p[2];
    }
    __syncthreads();
    float cx = cc[0], cy = cc[1], cz = cc[2];
    float bv = -1.f; int bi = 0x7fffffff;
    for (int i = tid; i < NPTS; i += 1024) {
      const float* p = xin + ((size_t)b * NPTS + i) * 6;
      float dx = p[0] - cx, dy = p[1] - cy, dz = p[2] - cz;
      float d = fminf(dist[i], dx * dx + dy * dy + dz * dz);
      dist[i] = d;
      if (d > bv) { bv = d; bi = i; }   // ascending i -> first max kept
    }
    rv[tid] = bv; ri[tid] = bi;
    __syncthreads();
    for (int s = 512; s > 0; s >>= 1) {
      if (tid < s) {
        float ov = rv[tid + s]; int oi = ri[tid + s];
        if (ov > rv[tid] || (ov == rv[tid] && oi < ri[tid])) { rv[tid] = ov; ri[tid] = oi; }
      }
      __syncthreads();
    }
    far = ri[0];
  }
  {
    int s = tid;
    int id = fps_idx[b * SPTS + s];
    const float* p = xin + ((size_t)b * NPTS + id) * 6;
    center[((size_t)b * SPTS + s) * 3 + 0] = p[0];
    center[((size_t)b * SPTS + s) * 3 + 1] = p[1];
    center[((size_t)b * SPTS + s) * 3 + 2] = p[2];
  }
}

// ---------------- KNN: 1 block per (s, b) ----------------
__global__ __launch_bounds__(128) void knn_kernel(
    const float* __restrict__ xin, const float* __restrict__ center,
    int* __restrict__ knn_idx, float* __restrict__ knn_d)
{
  int s = blockIdx.x, b = blockIdx.y, tid = threadIdx.x;
  __shared__ float d2[NPTS];
  __shared__ float rv[128];
  __shared__ int   ri[128];
  float cx = center[((size_t)(b * SPTS + s)) * 3 + 0];
  float cy = center[((size_t)(b * SPTS + s)) * 3 + 1];
  float cz = center[((size_t)(b * SPTS + s)) * 3 + 2];
  for (int i = tid; i < NPTS; i += 128) {
    const float* p = xin + ((size_t)b * NPTS + i) * 6;
    float dx = p[0] - cx, dy = p[1] - cy, dz = p[2] - cz;
    d2[i] = dx * dx + dy * dy + dz * dz;
  }
  __syncthreads();
  for (int j = 0; j < KNN; ++j) {
    float bv = 3e38f; int bi = 0x7fffffff;
    for (int i = tid; i < NPTS; i += 128) {
      float v = d2[i];
      if (v < bv) { bv = v; bi = i; }
    }
    rv[tid] = bv; ri[tid] = bi;
    __syncthreads();
    for (int st = 64; st > 0; st >>= 1) {
      if (tid < st) {
        float ov = rv[tid + st]; int oi = ri[tid + st];
        if (ov < rv[tid] || (ov == rv[tid] && oi < ri[tid])) { rv[tid] = ov; ri[tid] = oi; }
      }
      __syncthreads();
    }
    if (tid == 0) {
      knn_idx[((size_t)(b * SPTS + s)) * KNN + j] = ri[0];
      knn_d [((size_t)(b * SPTS + s)) * KNN + j] = rv[0];
      d2[ri[0]] = 3e38f;
    }
    __syncthreads();
  }
}

// ---------------- pos MLP + gather -> feat (f32 into out[...,128:256], f16 into flat16) ----
__global__ __launch_bounds__(256) void pos_feat_kernel(
    const float* __restrict__ xin, const float* __restrict__ center,
    const int* __restrict__ knn_idx, const float* __restrict__ knn_d,
    const float* __restrict__ points,
    const float* __restrict__ w1, const float* __restrict__ b1,
    const float* __restrict__ w2, const float* __restrict__ b2,
    const float* __restrict__ g1, const float* __restrict__ bb1,
    const float* __restrict__ m1, const float* __restrict__ vv1,
    const float* __restrict__ g2, const float* __restrict__ bb2,
    const float* __restrict__ m2, const float* __restrict__ vv2,
    float* __restrict__ out, _Float16* __restrict__ flat16)
{
  int s = blockIdx.x, b = blockIdx.y, tid = threadIdx.x;
  __shared__ float pin[KNN][10];
  __shared__ float hid[KNN][64];
  size_t gbase = (size_t)(b * SPTS + s) * KNN;
  if (tid < KNN) {
    int j = tid;
    int gi = knn_idx[gbase + j];
    const float* gp = xin + ((size_t)b * NPTS + gi) * 6;
    float cx = center[((size_t)(b * SPTS + s)) * 3 + 0];
    float cy = center[((size_t)(b * SPTS + s)) * 3 + 1];
    float cz = center[((size_t)(b * SPTS + s)) * 3 + 2];
    float gx = gp[0], gy = gp[1], gz = gp[2];
    pin[j][0] = cx; pin[j][1] = cy; pin[j][2] = cz;
    pin[j][3] = gx; pin[j][4] = gy; pin[j][5] = gz;
    pin[j][6] = gx - cx; pin[j][7] = gy - cy; pin[j][8] = gz - cz;
    pin[j][9] = knn_d[gbase + j];
  }
  __syncthreads();
  {
    int j = tid >> 3, h0 = (tid & 7) * 8;
    for (int h = h0; h < h0 + 8; ++h) {
      float a = b1[h];
#pragma unroll
      for (int q = 0; q < 10; ++q) a += w1[h * 10 + q] * pin[j][q];
      a = (a - m1[h]) * rsqrtf(vv1[h] + EPS_BN) * g1[h] + bb1[h];
      hid[j][h] = fmaxf(a, 0.f);
    }
  }
  __syncthreads();
  {
    int j = tid >> 3, d0 = (tid & 7) * 16;
    int gi = knn_idx[gbase + j];
    const float* pp = points + ((size_t)b * NPTS + gi) * DE;
    for (int d = d0; d < d0 + 16; ++d) {
      float a = b2[d];
      for (int q = 0; q < 64; ++q) a += w2[d * 64 + q] * hid[j][q];
      a = (a - m2[d]) * rsqrtf(vv2[d] + EPS_BN) * g2[d] + bb2[d];
      float f = fmaxf(a, 0.f) + pp[d];
      out[(gbase + j) * 256 + 128 + d] = f;                                 // feat f32 (grp half)
      flat16[((size_t)(b * SPTS + s)) * FLATF + j * DE + d] = (_Float16)f;  // feat f16
    }
  }
}

// ---------------- local neighborhood attention (WMMA), writes out[...,0:128] -------------
__global__ __launch_bounds__(128) void nei_kernel(
    const float* __restrict__ qw, const float* __restrict__ qb,
    const float* __restrict__ kw, const float* __restrict__ kbias,
    const _Float16* __restrict__ vwh, const float* __restrict__ vb,
    const float* __restrict__ gamma_nei, float* __restrict__ out)
{
  int s = blockIdx.x, b = blockIdx.y, tid = threadIdx.x;
  int lane = tid & 31, w = tid >> 5;
  __shared__ _Float16 fh[KNN][DE + 8];
  __shared__ _Float16 vh[KNN][DE + 8];
  __shared__ float qs[KNN][16], ksm[KNN][16];
  __shared__ _Float16 attnh[KNN][KNN];
  size_t obase = (size_t)(b * SPTS + s) * KNN * 256;

  for (int i = tid; i < KNN * DE; i += 128) {
    int m = i >> 7, d = i & 127;
    fh[m][d] = (_Float16)out[obase + (size_t)m * 256 + 128 + d];
  }
  __syncthreads();

  // q,k projections (K=16 too small for WMMA shape) — scalar
#pragma unroll
  for (int u = 0; u < 4; ++u) {
    int e = tid * 4 + u;
    int i = e >> 4, c = e & 15;
    float aq = qb[c], ak = kbias[c];
    for (int t = 0; t < DE; ++t) {
      float f = (float)fh[i][t];
      aq += qw[c * DE + t] * f;
      ak += kw[c * DE + t] * f;
    }
    qs[i][c] = aq; ksm[i][c] = ak;
  }

  // v = feat @ v_w^T + v_b via WMMA (M=32, N=128, K=128)
  {
    v8f acc[2][2] = {};
    for (int kk = 0; kk < 4; ++kk) {
      int k0 = kk * 32;
      v16h a0 = frag_ld_a(&fh[0][0], DE + 8, 0, k0);
      v16h a1 = frag_ld_a(&fh[0][0], DE + 8, 16, k0);
#pragma unroll
      for (int jn = 0; jn < 2; ++jn) {
        int nt = 2 * w + jn;
        v16h bf = frag_ld_bt(vwh, DE, nt * 16, k0);
        acc[0][jn] = wmma32(a0, bf, acc[0][jn]);
        acc[1][jn] = wmma32(a1, bf, acc[1][jn]);
      }
    }
#pragma unroll
    for (int jn = 0; jn < 2; ++jn) {
      int n = (2 * w + jn) * 16 + (lane & 15);
      int mb = (lane >> 4) * 8;
      float bias = vb[n];
#pragma unroll
      for (int mt = 0; mt < 2; ++mt)
#pragma unroll
        for (int r = 0; r < 8; ++r)
          vh[mt * 16 + mb + r][n] = (_Float16)(acc[mt][jn][r] + bias);
    }
  }
  __syncthreads();

  // scores + row softmax (32x32), one thread per row
  if (tid < KNN) {
    float sc[KNN]; float mx = -1e30f;
    for (int j = 0; j < KNN; ++j) {
      float a = 0.f;
#pragma unroll
      for (int e = 0; e < 16; ++e) a += qs[tid][e] * ksm[j][e];
      sc[j] = a; mx = fmaxf(mx, a);
    }
    float sum = 0.f;
    for (int j = 0; j < KNN; ++j) { float ev = __expf(sc[j] - mx); sc[j] = ev; sum += ev; }
    float inv = 1.f / sum;
    for (int j = 0; j < KNN; ++j) attnh[tid][j] = (_Float16)(sc[j] * inv);
  }
  __syncthreads();

  // nei = gamma * attn @ v + feat via WMMA (M=32, N=128, K=32)
  float gn = gamma_nei[0];
  {
    v8f acc[2][2] = {};
    v16h a0 = frag_ld_a(&attnh[0][0], KNN, 0, 0);
    v16h a1 = frag_ld_a(&attnh[0][0], KNN, 16, 0);
#pragma unroll
    for (int jn = 0; jn < 2; ++jn) {
      int nt = 2 * w + jn;
      v16h bf = frag_ld_b(&vh[0][0], DE + 8, nt * 16, 0);
      acc[0][jn] = wmma32(a0, bf, acc[0][jn]);
      acc[1][jn] = wmma32(a1, bf, acc[1][jn]);
    }
#pragma unroll
    for (int jn = 0; jn < 2; ++jn) {
      int n = (2 * w + jn) * 16 + (lane & 15);
      int mb = (lane >> 4) * 8;
#pragma unroll
      for (int mt = 0; mt < 2; ++mt)
#pragma unroll
        for (int r = 0; r < 8; ++r) {
          int m = mt * 16 + mb + r;
          size_t a_ = obase + (size_t)m * 256;
          out[a_ + n] = gn * acc[mt][jn][r] + out[a_ + 128 + n];
        }
    }
  }
}

// ------ group attention GEMM 1: S = flat @ flat^T, 128x64 tile, async double-buffered ------
__global__ __launch_bounds__(256) void gemm_s_kernel(
    const _Float16* __restrict__ flat16, float* __restrict__ S)
{
  int nt = blockIdx.x, mt = blockIdx.y, b = blockIdx.z;
  int tid = threadIdx.x, lane = tid & 31, w = tid >> 5;
  const _Float16* A = flat16 + (size_t)b * SPTS * FLATF;
  float* Sb = S + (size_t)b * SPTS * SPTS;
  __shared__ _Float16 At[2][128][40];
  __shared__ _Float16 Bt[2][64][40];
  int m0 = mt * 128, n0 = nt * 64;

  auto load_tiles = [&](int bsel, int k0) {
#pragma unroll
    for (int rep = 0; rep < 2; ++rep) {            // A: 128x32 = 512 b128 chunks
      int id = tid + rep * 256;
      int r_ = id >> 2, c8 = (id & 3) * 8;
      async_b128(&At[bsel][r_][c8], A + (size_t)(m0 + r_) * FLATF + k0 + c8);
    }
    int r_ = tid >> 2, c8 = (tid & 3) * 8;          // B: 64x32 = 256 b128 chunks
    async_b128(&Bt[bsel][r_][c8], A + (size_t)(n0 + r_) * FLATF + k0 + c8);
  };

  v8f acc[4] = {};
  load_tiles(0, 0);
  wait_async0();
  __syncthreads();
  int buf = 0;
  for (int k0 = 0; k0 < FLATF; k0 += 32) {
    if (k0 + 32 < FLATF) load_tiles(buf ^ 1, k0 + 32);   // overlap copy with WMMA
    v16h af = frag_ld_a(&At[buf][0][0], 40, w * 16, 0);
#pragma unroll
    for (int j = 0; j < 4; ++j) {
      v16h bf = frag_ld_bt(&Bt[buf][0][0], 40, j * 16, 0);
      acc[j] = wmma32(af, bf, acc[j]);
    }
    wait_async0();
    __syncthreads();
    buf ^= 1;
  }
#pragma unroll
  for (int j = 0; j < 4; ++j) {
    int n = n0 + j * 16 + (lane & 15);
    int mb = (lane >> 4) * 8;
#pragma unroll
    for (int r = 0; r < 8; ++r)
      Sb[(size_t)(m0 + w * 16 + mb + r) * SPTS + n] = acc[j][r];
  }
}

// ---------------- row softmax of S -> P (f16) ----------------
__global__ __launch_bounds__(256) void softmax_kernel(
    const float* __restrict__ S, _Float16* __restrict__ P)
{
  int m = blockIdx.x, b = blockIdx.y, tid = threadIdx.x;
  const float* row = S + ((size_t)(b * SPTS + m)) * SPTS;
  _Float16* prow = P + ((size_t)(b * SPTS + m)) * SPTS;
  __shared__ float red[256];
  float mx = -1e30f;
#pragma unroll
  for (int i = 0; i < 4; ++i) mx = fmaxf(mx, row[tid + i * 256]);
  red[tid] = mx; __syncthreads();
  for (int s = 128; s > 0; s >>= 1) {
    if (tid < s) red[tid] = fmaxf(red[tid], red[tid + s]);
    __syncthreads();
  }
  mx = red[0]; __syncthreads();
  float vals[4]; float sum = 0.f;
#pragma unroll
  for (int i = 0; i < 4; ++i) { vals[i] = __expf(row[tid + i * 256] - mx); sum += vals[i]; }
  red[tid] = sum; __syncthreads();
  for (int s = 128; s > 0; s >>= 1) {
    if (tid < s) red[tid] += red[tid + s];
    __syncthreads();
  }
  float inv = 1.f / red[0];
#pragma unroll
  for (int i = 0; i < 4; ++i) prow[tid + i * 256] = (_Float16)(vals[i] * inv);
}

// ---- group attention GEMM 2: out2 = gamma*(P @ flat) + feat, async double-buffered ----
__global__ __launch_bounds__(256) void gemm_g_kernel(
    const _Float16* __restrict__ P, const _Float16* __restrict__ flat16,
    const float* __restrict__ gamma_grp, float* __restrict__ out)
{
  int nt = blockIdx.x, mt = blockIdx.y, b = blockIdx.z;
  int tid = threadIdx.x, lane = tid & 31, w = tid >> 5;
  const _Float16* Ap = P + (size_t)b * SPTS * SPTS;
  const _Float16* Bm = flat16 + (size_t)b * SPTS * FLATF;
  __shared__ _Float16 At[2][128][40];
  __shared__ _Float16 Bt[2][32][72];
  int m0 = mt * 128, n0 = nt * 64;

  auto load_tiles = [&](int bsel, int k0) {
#pragma unroll
    for (int rep = 0; rep < 2; ++rep) {            // A: 128x32 halves
      int id = tid + rep * 256;
      int r_ = id >> 2, c8 = (id & 3) * 8;
      async_b128(&At[bsel][r_][c8], Ap + (size_t)(m0 + r_) * SPTS + k0 + c8);
    }
    int kr = tid >> 3, c8b = (tid & 7) * 8;         // B: 32(K) x 64(N) halves
    async_b128(&Bt[bsel][kr][c8b], Bm + (size_t)(k0 + kr) * FLATF + n0 + c8b);
  };

  v8f acc[4] = {};
  load_tiles(0, 0);
  wait_async0();
  __syncthreads();
  int buf = 0;
  for (int k0 = 0; k0 < SPTS; k0 += 32) {
    if (k0 + 32 < SPTS) load_tiles(buf ^ 1, k0 + 32);
    v16h af = frag_ld_a(&At[buf][0][0], 40, w * 16, 0);
#pragma unroll
    for (int j = 0; j < 4; ++j) {
      v16h bf = frag_ld_b(&Bt[buf][0][0], 72, j * 16, 0);
      acc[j] = wmma32(af, bf, acc[j]);
    }
    wait_async0();
    __syncthreads();
    buf ^= 1;
  }
  float g = gamma_grp[0];
#pragma unroll
  for (int j = 0; j < 4; ++j) {
    int n = n0 + j * 16 + (lane & 15);
    int mb = (lane >> 4) * 8;
#pragma unroll
    for (int r = 0; r < 8; ++r) {
      int m = m0 + w * 16 + mb + r;
      size_t a_ = ((size_t)(b * SPTS + m) * KNN + (n >> 7)) * 256 + 128 + (n & 127);
      out[a_] = g * acc[j][r] + out[a_];   // out2 currently holds feat -> residual
    }
  }
}

// ---------------- launch ----------------
extern "C" void kernel_launch(void* const* d_in, const int* in_sizes, int n_in,
                              void* d_out, int out_size, void* d_ws, size_t ws_size,
                              hipStream_t stream) {
  (void)in_sizes; (void)n_in; (void)out_size; (void)ws_size;
  const float* xyz_norm = (const float*)d_in[0];
  const float* emb_w1 = (const float*)d_in[1];
  const float* emb_b1 = (const float*)d_in[2];
  const float* emb_w2 = (const float*)d_in[3];
  const float* emb_b2 = (const float*)d_in[4];
  const float* emb_g1 = (const float*)d_in[5];
  const float* emb_bb1 = (const float*)d_in[6];
  const float* emb_m1 = (const float*)d_in[7];
  const float* emb_v1 = (const float*)d_in[8];
  const float* emb_g2 = (const float*)d_in[9];
  const float* emb_bb2 = (const float*)d_in[10];
  const float* emb_m2 = (const float*)d_in[11];
  const float* emb_v2 = (const float*)d_in[12];
  const float* pos_w1 = (const float*)d_in[13];
  const float* pos_b1 = (const float*)d_in[14];
  const float* pos_w2 = (const float*)d_in[15];
  const float* pos_b2 = (const float*)d_in[16];
  const float* pos_g1 = (const float*)d_in[17];
  const float* pos_bb1 = (const float*)d_in[18];
  const float* pos_m1 = (const float*)d_in[19];
  const float* pos_v1 = (const float*)d_in[20];
  const float* pos_g2 = (const float*)d_in[21];
  const float* pos_bb2 = (const float*)d_in[22];
  const float* pos_m2 = (const float*)d_in[23];
  const float* pos_v2 = (const float*)d_in[24];
  const float* q_w = (const float*)d_in[25];
  const float* q_b = (const float*)d_in[26];
  const float* k_w = (const float*)d_in[27];
  const float* k_b = (const float*)d_in[28];
  const float* v_w = (const float*)d_in[29];
  const float* v_b = (const float*)d_in[30];
  const float* gamma_nei = (const float*)d_in[31];
  const float* gamma_grp = (const float*)d_in[32];
  float* out = (float*)d_out;

  char* wp = (char*)d_ws;
  auto take = [&](size_t bytes) { char* p = wp; wp += (bytes + 255) & ~(size_t)255; return p; };
  float*    points  = (float*)   take((size_t)BATCH * NPTS * DE * 4);
  _Float16* flat16  = (_Float16*)take((size_t)BATCH * SPTS * FLATF * 2);
  float*    Sbuf    = (float*)   take((size_t)BATCH * SPTS * SPTS * 4);
  _Float16* Pbuf    = (_Float16*)take((size_t)BATCH * SPTS * SPTS * 2);
  int*      fps_idx = (int*)     take((size_t)BATCH * SPTS * 4);
  float*    center  = (float*)   take((size_t)BATCH * SPTS * 3 * 4);
  int*      knn_idx = (int*)     take((size_t)BATCH * SPTS * KNN * 4);
  float*    knn_d   = (float*)   take((size_t)BATCH * SPTS * KNN * 4);
  _Float16* v_wh    = (_Float16*)take((size_t)DE * DE * 2);

  cvt_f16_kernel<<<dim3((DE * DE + 255) / 256), dim3(256), 0, stream>>>(v_w, v_wh, DE * DE);
  emb_kernel<<<dim3((BATCH * NPTS) / 256), dim3(256), 0, stream>>>(
      xyz_norm, emb_w1, emb_b1, emb_w2, emb_b2,
      emb_g1, emb_bb1, emb_m1, emb_v1, emb_g2, emb_bb2, emb_m2, emb_v2, points);
  fps_kernel<<<dim3(BATCH), dim3(1024), 0, stream>>>(xyz_norm, fps_idx, center);
  knn_kernel<<<dim3(SPTS, BATCH), dim3(128), 0, stream>>>(xyz_norm, center, knn_idx, knn_d);
  pos_feat_kernel<<<dim3(SPTS, BATCH), dim3(256), 0, stream>>>(
      xyz_norm, center, knn_idx, knn_d, points,
      pos_w1, pos_b1, pos_w2, pos_b2,
      pos_g1, pos_bb1, pos_m1, pos_v1, pos_g2, pos_bb2, pos_m2, pos_v2,
      out, flat16);
  nei_kernel<<<dim3(SPTS, BATCH), dim3(128), 0, stream>>>(
      q_w, q_b, k_w, k_b, v_wh, v_b, gamma_nei, out);
  gemm_s_kernel<<<dim3(SPTS / 64, SPTS / 128, BATCH), dim3(256), 0, stream>>>(flat16, Sbuf);
  softmax_kernel<<<dim3(SPTS, BATCH), dim3(256), 0, stream>>>(Sbuf, Pbuf);
  gemm_g_kernel<<<dim3(FLATF / 64, SPTS / 128, BATCH), dim3(256), 0, stream>>>(
      Pbuf, flat16, gamma_grp, out);
}
